// Router_38972533243957
// MI455X (gfx1250) — matrix operand order repaired and checked by reading, not compiled, not run on hardware
//
#include <hip/hip_runtime.h>
#include <hip/hip_bf16.h>

#define D_MODEL 2816
#define N_EXPERTS 128
#define TOP_K 8
#define KTILES (D_MODEL / 32)     // 88 k-tiles of 32
#define NTILES (N_EXPERTS / 16)   // 8 expert tiles of 16
#define WAVES 8
#define BLOCK (WAVES * 32)
#define TOKENS_TOTAL (4 * 4096)

typedef __attribute__((ext_vector_type(16))) __bf16 v16bf;
typedef __attribute__((ext_vector_type(8)))  float  v8f;

// ---------------------------------------------------------------------------
// Kernel 1: pack proj_w (fp32, [E=128][D=2816] row major) into bf16 B-fragments
// in workspace, folding scale[d] * D_MODEL^-0.5 into the weights.
// Flat layout: [kt][ntile][lane][16 bf16]  (one contiguous 32B chunk per lane)
// B (32x16, 16-bit): lanes 0-15 hold K=0..7,16..23, lanes 16-31 hold K=8..15,24..31,
// lane%16 = expert column within the tile, element i = K (i<8 ? g*8+i : 16+g*8+i-8).
// ---------------------------------------------------------------------------
__global__ void pack_b_kernel(const float* __restrict__ w,
                              const float* __restrict__ scale,
                              __bf16* __restrict__ bpack) {
  int gid = blockIdx.x * blockDim.x + threadIdx.x;
  if (gid >= KTILES * NTILES * 32 * 16) return;
  int i  = gid & 15;
  int l  = (gid >> 4) & 31;
  int n  = (gid >> 9) & 7;
  int kt = gid >> 12;
  int e = n * 16 + (l & 15);
  int g = l >> 4;
  int d = kt * 32 + (i < 8 ? g * 8 + i : 16 + g * 8 + (i - 8));
  float v = w[e * D_MODEL + d] * scale[d] * 0.018844442f; // 1/sqrt(2816)
  bpack[gid] = (__bf16)v;
}

// ---------------------------------------------------------------------------
// Kernel 2: one wave per 16 tokens. RMSNorm -> bf16 WMMA GEMM vs 128 experts
// -> top-8 (softmax denominator cancels under renormalization) -> scatter.
// ---------------------------------------------------------------------------
__global__ void __launch_bounds__(BLOCK) router_main_kernel(
    const float* __restrict__ x,
    const __bf16* __restrict__ bpack,
    const float* __restrict__ per_expert_scale,
    float* __restrict__ out) {
  __shared__ float smem[WAVES][16 * 128];   // 64 KB: reduction scratch + scores

  const int lane = threadIdx.x & 31;
  const int wave = threadIdx.x >> 5;
  const int g    = lane >> 4;      // lane group (K-half selector)
  const int c    = lane & 15;      // row (A) / column (B,C,D) within tile
  const long tok0 = (long)(blockIdx.x * WAVES + wave) * 16;

  // ---- zero this wave's 16 output rows (scatter overwrites later; same-wave
  //      stores to the same address are kept in order) ----
  float4 z4 = make_float4(0.f, 0.f, 0.f, 0.f);
  float4* out4 = (float4*)out;
  #pragma unroll
  for (int m = 0; m < 16; ++m)
    out4[(tok0 + m) * (N_EXPERTS / 4) + lane] = z4;

  // ---- phase 1: sum of squares per token (coalesced float4 streaming) ----
  float ss[16];
  #pragma unroll
  for (int m = 0; m < 16; ++m) ss[m] = 0.f;
  for (int it = 0; it < D_MODEL / 128; ++it) {
    const float4* xb = (const float4*)(x + tok0 * D_MODEL + it * 128) + lane;
    #pragma unroll
    for (int m = 0; m < 16; ++m) {
      float4 v = xb[m * (D_MODEL / 4)];
      ss[m] += v.x * v.x + v.y * v.y + v.z * v.z + v.w * v.w;
    }
  }
  float* sw = &smem[wave][0];
  #pragma unroll
  for (int m = 0; m < 16; ++m) sw[lane * 16 + m] = ss[m];
  __syncthreads();
  float tot = 0.f;
  for (int j = 0; j < 32; ++j) tot += sw[j * 16 + c];
  const float myrms = rsqrtf(tot * (1.0f / D_MODEL) + 1e-6f);
  __syncthreads();

  // ---- phase 2: GEMM, K-loop of 88 x (8 x v_wmma_f32_16x16x32_bf16) ----
  v8f acc[NTILES];
  #pragma unroll
  for (int n = 0; n < NTILES; ++n)
    acc[n] = (v8f){0.f, 0.f, 0.f, 0.f, 0.f, 0.f, 0.f, 0.f};

  const float* arow = x + (tok0 + c) * D_MODEL;
  const v16bf* bp = (const v16bf*)bpack;
  for (int kt = 0; kt < KTILES; ++kt) {
    // 16-bit A 16x32 layout: group 0 -> K in [0..7] U [16..23],
    //                        group 1 -> K in [8..15] U [24..31]
    const float* ab = arow + kt * 32 + g * 8;
    float4 f0 = *(const float4*)(ab);
    float4 f1 = *(const float4*)(ab + 4);
    float4 f2 = *(const float4*)(ab + 16);
    float4 f3 = *(const float4*)(ab + 20);
    v16bf a;
    a[0]  = (__bf16)(f0.x * myrms); a[1]  = (__bf16)(f0.y * myrms);
    a[2]  = (__bf16)(f0.z * myrms); a[3]  = (__bf16)(f0.w * myrms);
    a[4]  = (__bf16)(f1.x * myrms); a[5]  = (__bf16)(f1.y * myrms);
    a[6]  = (__bf16)(f1.z * myrms); a[7]  = (__bf16)(f1.w * myrms);
    a[8]  = (__bf16)(f2.x * myrms); a[9]  = (__bf16)(f2.y * myrms);
    a[10] = (__bf16)(f2.z * myrms); a[11] = (__bf16)(f2.w * myrms);
    a[12] = (__bf16)(f3.x * myrms); a[13] = (__bf16)(f3.y * myrms);
    a[14] = (__bf16)(f3.z * myrms); a[15] = (__bf16)(f3.w * myrms);
    #pragma unroll
    for (int n = 0; n < NTILES; ++n) {
      v16bf b = bp[(kt * NTILES + n) * 32 + lane];
      acc[n] = __builtin_amdgcn_wmma_f32_16x16x32_bf16(
          false, a, false, b, (short)0, acc[n], false, false);
    }
  }

  // ---- scores -> LDS (C/D layout: VGPR r, row = r + 8*g, col = ntile*16+c) --
  #pragma unroll
  for (int n = 0; n < NTILES; ++n) {
    #pragma unroll
    for (int r = 0; r < 8; ++r) {
      sw[(r + 8 * g) * 128 + n * 16 + c] = acc[n][r];
    }
  }
  __syncthreads();

  // ---- top-8, exp/renormalize (softmax denom cancels), scatter ----
  if (lane < 16) {
    float* row = sw + lane * 128;
    float vals[TOP_K];
    int   idxs[TOP_K];
    #pragma unroll
    for (int k = 0; k < TOP_K; ++k) {
      float mx = -1e30f; int mi = 0;
      #pragma unroll 4
      for (int e = 0; e < N_EXPERTS; ++e) {
        float v = row[e];
        if (v > mx) { mx = v; mi = e; }
      }
      vals[k] = mx; idxs[k] = mi;
      row[mi] = -1e30f;
    }
    float m0 = vals[0];
    float w[TOP_K];
    float sum = 0.f;
    #pragma unroll
    for (int k = 0; k < TOP_K; ++k) { w[k] = __expf(vals[k] - m0); sum += w[k]; }
    float inv = 1.f / sum;
    float* orow = out + (size_t)(tok0 + lane) * N_EXPERTS;
    #pragma unroll
    for (int k = 0; k < TOP_K; ++k)
      orow[idxs[k]] = w[k] * inv * per_expert_scale[idxs[k]];
  }
}

extern "C" void kernel_launch(void* const* d_in, const int* in_sizes, int n_in,
                              void* d_out, int out_size, void* d_ws, size_t ws_size,
                              hipStream_t stream) {
  (void)in_sizes; (void)n_in; (void)out_size; (void)ws_size;
  const float* x      = (const float*)d_in[0];   // (4,4096,2816) f32
  const float* proj_w = (const float*)d_in[1];   // (128,2816) f32
  const float* scale  = (const float*)d_in[2];   // (2816,) f32
  const float* pes    = (const float*)d_in[3];   // (128,) f32
  float* out = (float*)d_out;                    // (4,4096,128) f32
  __bf16* bpack = (__bf16*)d_ws;                 // 704 KB packed bf16 B

  int nvals = KTILES * NTILES * 32 * 16;
  pack_b_kernel<<<(nvals + 255) / 256, 256, 0, stream>>>(proj_w, scale, bpack);

  int nblocks = TOKENS_TOTAL / (WAVES * 16);     // 128 blocks x 8 waves x 16 tokens
  router_main_kernel<<<nblocks, BLOCK, 0, stream>>>(x, bpack, pes, out);
}
